// DeepSeekV31Attention_68255620268082
// MI455X (gfx1250) — compile-verified, matrix operand order checked by reading, not run on hardware
//
#include <hip/hip_runtime.h>
#include <hip/hip_bf16.h>

// ---------------------------------------------------------------------------
// DeepSeek V3.1 MLA attention for MI455X (gfx1250, wave32).
// All matmuls run on v_wmma_f32_16x16x32_bf16 (f32 accum, bf16 operands).
// Fragments: 2x ds_load_b128 (K-contiguous LDS layouts, 16B-aligned rows).
// GEMM: 128x128 tile, ping-pong LDS (1 barrier/K-step), strength-reduced
// addressing. Flash: split-K score phase keeps all 8 waves on the WMMA pipe.
// ---------------------------------------------------------------------------

typedef __attribute__((ext_vector_type(16))) __bf16          v16bf;
typedef __attribute__((ext_vector_type(8)))  float           v8f;

__device__ __forceinline__ unsigned short f2bf(float f) {
  // round-to-nearest-even f32 -> bf16
  unsigned int u = __float_as_uint(f);
  unsigned int r = u + 0x7FFFu + ((u >> 16) & 1u);
  return (unsigned short)(r >> 16);
}

// Pack 4 floats -> 4 bf16, one 8-byte LDS store (p must be 8B aligned).
__device__ __forceinline__ void st_bf4(unsigned short* p, float4 v) {
  uint2 u;
  u.x = (unsigned int)f2bf(v.x) | ((unsigned int)f2bf(v.y) << 16);
  u.y = (unsigned int)f2bf(v.z) | ((unsigned int)f2bf(v.w) << 16);
  *(uint2*)p = u;
}

// One lane's 16x32 WMMA fragment: two contiguous 8-element bf16 runs.
// lanes 0-15: K = {0..7, 16..23}; lanes 16-31: K = {8..15, 24..31}.
struct FragU { uint4 lo, hi; };
__device__ __forceinline__ v16bf ld_frag16(const unsigned short* row, int lane) {
  const unsigned short* p = row + ((lane >> 4) << 3);
  FragU f;
  f.lo = *(const uint4*)(p);        // ds_load_b128
  f.hi = *(const uint4*)(p + 16);   // ds_load_b128
  return __builtin_bit_cast(v16bf, f);
}

__device__ __forceinline__ v8f wmma_bf16(v16bf a, v16bf b, v8f c) {
  return __builtin_amdgcn_wmma_f32_16x16x32_bf16(false, a, false, b, (short)0, c,
                                                 false, false);
}

__device__ __forceinline__ v8f vzero8() {
  return (v8f){0.f, 0.f, 0.f, 0.f, 0.f, 0.f, 0.f, 0.f};
}

// ---------------------------------------------------------------------------
// Batched WMMA GEMM:  C[z] = A[z] (MxK, f32) * B[z] (KxN or NxK, f32)
// ---------------------------------------------------------------------------
template <bool TRANS_B, bool OUT_BF16>
__global__ __launch_bounds__(256) void gemm_wmma(
    const float* __restrict__ A, long sA, int lda,
    const float* __restrict__ B, long sB, int ldb,
    void* __restrict__ Cv, long sC, int ldc,
    int M, int N, int K) {
  constexpr int LDT = 40;                    // 80B rows: 16B aligned, conflict-free
  __shared__ unsigned short As[2][128][LDT]; // [buf][m][k]
  __shared__ unsigned short Bt[2][128][LDT]; // [buf][n][k] (K-contiguous)

  const int tid  = threadIdx.x;
  const int lane = tid & 31;
  const int wv   = tid >> 5;   // 8 waves
  const int wm   = wv >> 1;    // 0..3 -> 32-row strip
  const int wn   = wv & 1;     // 0..1 -> 64-col strip
  const int bn0  = blockIdx.x * 128;
  const int bm0  = blockIdx.y * 128;

  A += (long)blockIdx.z * sA;
  B += (long)blockIdx.z * sB;
  float*          Cf = (float*)Cv + (long)blockIdx.z * sC;
  unsigned short* Ch = (unsigned short*)Cv + (long)blockIdx.z * sC;

  v8f acc[2][4];
#pragma unroll
  for (int mt = 0; mt < 2; ++mt)
#pragma unroll
    for (int nt = 0; nt < 4; ++nt) acc[mt][nt] = vzero8();

  // ---- loop-invariant per-thread decomposition ----
  const int arow = tid >> 3;           // 0..31
  const int ac4  = (tid & 7) << 2;     // 0..28
  const float* pA  = A + (long)(bm0 + arow) * lda + ac4;
  const float* pBt = B + (long)(bn0 + arow) * ldb + ac4;   // TRANS_B layout
  bool okBt[4];
#pragma unroll
  for (int i = 0; i < 4; ++i) okBt[i] = (bn0 + i * 32 + arow) < N;
  const int bn   = tid & 127;
  const int bkg  = (tid >> 7) << 2;    // 0 or 4
  const float* pBn = B + (long)bkg * ldb + (bn0 + bn);     // non-trans layout
  const bool okBn  = (bn0 + bn) < N;

  float4 ra[4], rb[4];

  auto load_tile = [&]() {
#pragma unroll
    for (int i = 0; i < 4; ++i)
      ra[i] = *(const float4*)(pA + (long)(i * 32) * lda);
    pA += 32;
    if (TRANS_B) {
#pragma unroll
      for (int i = 0; i < 4; ++i)
        rb[i] = okBt[i] ? *(const float4*)(pBt + (long)(i * 32) * ldb)
                        : make_float4(0.f, 0.f, 0.f, 0.f);
      pBt += 32;
    } else {
#pragma unroll
      for (int i = 0; i < 4; ++i) {
        if (okBn) {
          rb[i].x = pBn[(long)(i * 8 + 0) * ldb];
          rb[i].y = pBn[(long)(i * 8 + 1) * ldb];
          rb[i].z = pBn[(long)(i * 8 + 2) * ldb];
          rb[i].w = pBn[(long)(i * 8 + 3) * ldb];
        } else {
          rb[i] = make_float4(0.f, 0.f, 0.f, 0.f);
        }
      }
      pBn += (long)32 * ldb;
    }
  };

  auto store_tile = [&](int b) {
#pragma unroll
    for (int i = 0; i < 4; ++i) st_bf4(&As[b][i * 32 + arow][ac4], ra[i]);
    if (TRANS_B) {
#pragma unroll
      for (int i = 0; i < 4; ++i) st_bf4(&Bt[b][i * 32 + arow][ac4], rb[i]);
    } else {
#pragma unroll
      for (int i = 0; i < 4; ++i) st_bf4(&Bt[b][bn][bkg + i * 8], rb[i]);
    }
  };

  const int frow = lane & 15;
  auto compute = [&](int b) {
    v16bf a0 = ld_frag16(&As[b][wm * 32 + frow][0], lane);
    v16bf a1 = ld_frag16(&As[b][wm * 32 + 16 + frow][0], lane);
#pragma unroll
    for (int nt = 0; nt < 4; ++nt) {
      v16bf bb = ld_frag16(&Bt[b][wn * 64 + nt * 16 + frow][0], lane);
      acc[0][nt] = wmma_bf16(a0, bb, acc[0][nt]);
      acc[1][nt] = wmma_bf16(a1, bb, acc[1][nt]);
    }
  };

  // ---- ping-pong K pipeline: 1 barrier per K-step ----
  load_tile();
  store_tile(0);
  __syncthreads();
  int cur = 0;
  for (int k0 = 32; k0 < K; k0 += 32) {
    load_tile();           // global loads for tile k0
    compute(cur);          // WMMAs on tile k0-32
    store_tile(cur ^ 1);   // write tile k0 into the other buffer
    __syncthreads();
    cur ^= 1;
  }
  compute(cur);

  // ---- epilogue ----
  const int hi8  = (lane >> 4) << 3;
  const int col0 = bn0 + wn * 64 + frow;
#pragma unroll
  for (int mt = 0; mt < 2; ++mt)
#pragma unroll
    for (int nt = 0; nt < 4; ++nt) {
      const int col = col0 + nt * 16;
      if (col < N) {
#pragma unroll
        for (int v = 0; v < 8; ++v) {
          int row = bm0 + wm * 32 + mt * 16 + v + hi8;
          if (OUT_BF16) Ch[(long)row * ldc + col] = f2bf(acc[mt][nt][v]);
          else          Cf[(long)row * ldc + col] = acc[mt][nt][v];
        }
      }
    }
}

// ---------------------------------------------------------------------------
// YaRN RoPE, in place on x of shape (S, Hh, 64).
// ---------------------------------------------------------------------------
__global__ __launch_bounds__(256) void rope_kernel(float* __restrict__ x, int S, int Hh) {
  int idx = blockIdx.x * blockDim.x + threadIdx.x;
  int total = S * Hh * 32;
  if (idx >= total) return;
  int i = idx & 31;
  int h = (idx >> 5) % Hh;
  int s = idx / (Hh * 32);

  float freq = __expf(-((float)(2 * i) / 64.f) * __logf(10000.f));
  float wl   = 6.283185307179586f / freq;
  float ramp = (wl / 4096.f - 1.f) / (32.f - 1.f);
  ramp       = fminf(fmaxf(ramp, 0.f), 1.f);
  float invf = freq / (1.f - ramp + ramp * 40.f);
  float th   = (float)s * invf;
  float c = cosf(th), sn = sinf(th);

  float* p  = x + ((long)s * Hh + h) * 64;
  float x1 = p[i], x2 = p[i + 32];
  p[i]      = x1 * c - x2 * sn;
  p[i + 32] = x1 * sn + x2 * c;
}

// ---------------------------------------------------------------------------
// Flash MLA attention (absorbed latent form):
//   scores[s,t] = q_lat[s]·c_kv[t] + q_rope[s]·k_rope[t]  (causal softmax)
//   out_lat     = P @ c_kv
// Block: 32 query rows of one head, 8 waves, key tiles of 32.
// Score phase is split-K: waves 0-3 contract r=[0,256)+rope[0,32),
// waves 4-7 contract r=[256,512)+rope[32,64) into separate partial buffers,
// so ALL waves issue 9 score WMMAs + 8 O-update WMMAs per K-tile.
// ---------------------------------------------------------------------------
__global__ __launch_bounds__(256) void flash_mla(
    const unsigned short* __restrict__ qlat,  // (H, S, 512) bf16
    const float* __restrict__ qrope,          // (S, H*64)  f32 (RoPE applied)
    const float* __restrict__ ckv,            // (S, 512)   f32
    const float* __restrict__ krope,          // (S, 64)    f32 (RoPE applied)
    float* __restrict__ olat,                 // (H, S, 512) f32
    int S, int ldq) {
  __shared__ unsigned short Ql[32][520];  // [s][r]
  __shared__ unsigned short Qr[32][72];   // [s][r_rope]
  __shared__ unsigned short Kc[32][520];  // [t][r]  (scores: K=r contiguous)
  __shared__ unsigned short KT[512][40];  // [r][t]  (O-update: K=t contiguous)
  __shared__ unsigned short Kr[32][72];   // [t][r_rope]
  __shared__ float          Sc[2][32][32];// partial scores per wave-group
  __shared__ unsigned short Pp[32][40];   // [s][t] bf16 probs
  __shared__ float mrow[32], lrow[32], arow[32];

  const int tid = threadIdx.x, lane = tid & 31, wv = tid >> 5;
  const int h   = blockIdx.y;
  const int sq0 = blockIdx.x * 32;
  const float scale = 0.07216878364870323f;  // 1/sqrt(DN+DR) = 1/sqrt(192)

  // ---- load Q tiles once ----
  const unsigned short* qh = qlat + ((long)h * S + sq0) * 512;
  for (int i = tid; i < 32 * 128; i += 256) {           // 8B chunks of bf16
    int r = i >> 7, c4 = (i & 127) << 2;
    *(uint2*)&Ql[r][c4] = *(const uint2*)&qh[(long)r * 512 + c4];
  }
  {
    const int r0 = tid >> 4, c4 = (tid & 15) << 2;      // 2 chunks per thread
    const float* pQ = qrope + (long)(sq0 + r0) * ldq + h * 64 + c4;
#pragma unroll
    for (int j = 0; j < 2; ++j)
      st_bf4(&Qr[r0 + j * 16][c4], *(const float4*)(pQ + (long)(j * 16) * ldq));
  }
  if (tid < 32) { mrow[tid] = -1e30f; lrow[tid] = 0.f; arow[tid] = 0.f; }

  v8f o[2][4];
#pragma unroll
  for (int mt = 0; mt < 2; ++mt)
#pragma unroll
    for (int nt = 0; nt < 4; ++nt) o[mt][nt] = vzero8();

  __syncthreads();

  // ---- loop-invariant staging decomposition ----
  const int ct  = tid >> 7;            // 0..1
  const int cr4 = (tid & 127) << 2;    // 0..508
  const float* pC = ckv + (long)ct * 512 + cr4;    // += 32*512 per step
  unsigned short* dKc = &Kc[ct][cr4];
  unsigned short* dKT = &KT[cr4][ct];
  const int kt  = tid >> 4;            // 0..15
  const int kc4 = (tid & 15) << 2;
  const float* pR = krope + (long)kt * 64 + kc4;   // += 32*64 per step
  unsigned short* dKr = &Kr[kt][kc4];

  const int frow = lane & 15;
  const int hi8  = (lane >> 4) << 3;

  for (int t0 = 0; t0 <= sq0; t0 += 32) {
    // ---- stage K/V latent tile (f32 -> bf16, both orientations) ----
#pragma unroll
    for (int j = 0; j < 16; ++j) {     // rows j*2 + ct
      float4 v = *(const float4*)(pC + (long)j * 1024);
      st_bf4(dKc + j * (2 * 520), v);
      unsigned short* q = dKT + j * 2;
      q[0 * 40] = f2bf(v.x);
      q[1 * 40] = f2bf(v.y);
      q[2 * 40] = f2bf(v.z);
      q[3 * 40] = f2bf(v.w);
    }
    pC += 32 * 512;
#pragma unroll
    for (int j = 0; j < 2; ++j)
      st_bf4(dKr + j * (16 * 72), *(const float4*)(pR + (long)j * (16 * 64)));
    pR += 32 * 64;
    if (t0 + 32 <= sq0)
      __builtin_prefetch(pC + (long)(tid >> 3) * 512 + (tid & 7) * 64, 0, 1);
    __syncthreads();

    // ---- scores (split-K over wave groups, all 8 waves active) ----
    {
      const int g  = wv >> 2;          // contraction half
      const int id = wv & 3;           // which 16x16 score tile
      const int mt = id >> 1, nt = id & 1;
      const int mr = mt * 16 + frow;
      const int nr = nt * 16 + frow;
      v8f s = vzero8();
#pragma unroll
      for (int ks = 0; ks < 8; ++ks) {
        v16bf a = ld_frag16(&Ql[mr][(g * 8 + ks) * 32], lane);
        v16bf b = ld_frag16(&Kc[nr][(g * 8 + ks) * 32], lane);
        s = wmma_bf16(a, b, s);
      }
      {
        v16bf a = ld_frag16(&Qr[mr][g * 32], lane);
        v16bf b = ld_frag16(&Kr[nr][g * 32], lane);
        s = wmma_bf16(a, b, s);
      }
#pragma unroll
      for (int v = 0; v < 8; ++v)
        Sc[g][mt * 16 + v + hi8][nt * 16 + frow] = s[v] * scale;
    }
    __syncthreads();

    // ---- online softmax (one thread per query row) ----
    if (tid < 32) {
      const int r = tid;
      const int sg = sq0 + r;
      float mold = mrow[r];
      float mx = mold;
      for (int c = 0; c < 32; ++c)
        if (t0 + c <= sg) mx = fmaxf(mx, Sc[0][r][c] + Sc[1][r][c]);
      float al = __expf(mold - mx);
      float sum = 0.f;
      for (int c = 0; c < 32; ++c) {
        float p = (t0 + c <= sg) ? __expf(Sc[0][r][c] + Sc[1][r][c] - mx) : 0.f;
        Pp[r][c] = f2bf(p);
        sum += p;
      }
      lrow[r] = lrow[r] * al + sum;
      mrow[r] = mx;
      arow[r] = al;
    }
    __syncthreads();

    // ---- O = diag(alpha) O + P @ c_kv  (all 8 waves, 64 cols each) ----
#pragma unroll
    for (int mt = 0; mt < 2; ++mt) {
      v16bf a = ld_frag16(&Pp[mt * 16 + frow][0], lane);
#pragma unroll
      for (int nt = 0; nt < 4; ++nt) {
#pragma unroll
        for (int v = 0; v < 8; ++v)
          o[mt][nt][v] *= arow[mt * 16 + v + hi8];
        v16bf b = ld_frag16(&KT[wv * 64 + nt * 16 + frow][0], lane);
        o[mt][nt] = wmma_bf16(a, b, o[mt][nt]);
      }
    }
    __syncthreads();
  }

  // ---- epilogue: normalize by running sum, store out_lat ----
#pragma unroll
  for (int mt = 0; mt < 2; ++mt)
#pragma unroll
    for (int nt = 0; nt < 4; ++nt)
#pragma unroll
      for (int v = 0; v < 8; ++v) {
        int row = mt * 16 + v + hi8;
        int col = wv * 64 + nt * 16 + frow;
        olat[((long)h * S + sq0 + row) * 512 + col] = o[mt][nt][v] / lrow[row];
      }
}

// ---------------------------------------------------------------------------
// Host-side orchestration.
// ---------------------------------------------------------------------------
extern "C" void kernel_launch(void* const* d_in, const int* in_sizes, int n_in,
                              void* d_out, int out_size, void* d_ws, size_t ws_size,
                              hipStream_t stream) {
  const float* hidden      = (const float*)d_in[0];
  const float* w_q_down    = (const float*)d_in[1];
  const float* w_q_up_nope = (const float*)d_in[2];
  const float* w_q_up_rope = (const float*)d_in[3];
  const float* w_kv_down   = (const float*)d_in[4];
  const float* w_k_rope    = (const float*)d_in[5];
  const float* w_uk        = (const float*)d_in[6];
  const float* w_uv        = (const float*)d_in[7];
  const float* w_out       = (const float*)d_in[8];
  float* out = (float*)d_out;

  constexpr int S = 2048, D = 4096, H = 32, QL = 1536, KVL = 512;
  constexpr int DN = 128, DR = 64, DV = 128;

  char* ws = (char*)d_ws;
  auto alloc = [&](size_t bytes) {
    char* p = ws;
    ws += (bytes + 255) & ~(size_t)255;
    return (void*)p;
  };
  float* c_q      = (float*)alloc((size_t)S * QL * 4);
  float* q_nope   = (float*)alloc((size_t)S * H * DN * 4);
  float* q_rope   = (float*)alloc((size_t)S * H * DR * 4);
  float* c_kv     = (float*)alloc((size_t)S * KVL * 4);
  float* k_rope   = (float*)alloc((size_t)S * DR * 4);
  unsigned short* q_lat = (unsigned short*)alloc((size_t)H * S * KVL * 2);
  float* out_lat  = (float*)alloc((size_t)H * S * KVL * 4);
  float* attn_out = (float*)alloc((size_t)S * H * DV * 4);

  const dim3 blk(256);
  auto gx = [](int n) { return (n + 127) / 128; };

  gemm_wmma<false, false><<<dim3(gx(QL), S / 128, 1), blk, 0, stream>>>(
      hidden, 0, D, w_q_down, 0, QL, c_q, 0, QL, S, QL, D);
  gemm_wmma<false, false><<<dim3(gx(KVL), S / 128, 1), blk, 0, stream>>>(
      hidden, 0, D, w_kv_down, 0, KVL, c_kv, 0, KVL, S, KVL, D);
  gemm_wmma<false, false><<<dim3(gx(DR), S / 128, 1), blk, 0, stream>>>(
      hidden, 0, D, w_k_rope, 0, DR, k_rope, 0, DR, S, DR, D);
  gemm_wmma<false, false><<<dim3(gx(H * DN), S / 128, 1), blk, 0, stream>>>(
      c_q, 0, QL, w_q_up_nope, 0, H * DN, q_nope, 0, H * DN, S, H * DN, QL);
  gemm_wmma<false, false><<<dim3(gx(H * DR), S / 128, 1), blk, 0, stream>>>(
      c_q, 0, QL, w_q_up_rope, 0, H * DR, q_rope, 0, H * DR, S, H * DR, QL);

  rope_kernel<<<(S * H * 32 + 255) / 256, 256, 0, stream>>>(q_rope, S, H);
  rope_kernel<<<(S * 32 + 255) / 256, 256, 0, stream>>>(k_rope, S, 1);

  gemm_wmma<false, true><<<dim3(gx(KVL), S / 128, H), blk, 0, stream>>>(
      q_nope, DN, H * DN, w_uk, (long)DN * KVL, KVL,
      q_lat, (long)S * KVL, KVL, S, KVL, DN);

  flash_mla<<<dim3(S / 32, H), 256, 0, stream>>>(
      q_lat, q_rope, c_kv, k_rope, out_lat, S, H * DR);

  gemm_wmma<true, false><<<dim3(gx(DV), S / 128, H), blk, 0, stream>>>(
      out_lat, (long)S * KVL, KVL, w_uv, (long)DV * KVL, KVL,
      attn_out, DV, H * DV, S, DV, KVL);

  gemm_wmma<false, false><<<dim3(gx(D), S / 128, 1), blk, 0, stream>>>(
      attn_out, 0, H * DV, w_out, 0, D, out, 0, D, S, D, H * DV);
}
